// CrossLayerFeatureUpdate_20074677141961
// MI455X (gfx1250) — compile-verified, matrix operand order checked by reading, not run on hardware
//
#include <hip/hip_runtime.h>
#include <hip/hip_bf16.h>
#include <stdint.h>

// CDNA5 (gfx1250) fused "cross-layer feature update":
//   two dense GEMMs (A @ W^T) with trivial gather on A, fused LayerNorm+ReLU.
// bf16x3 WMMA (v_wmma_f32_16x16x32_bf16) with f32 accumulate for ~fp32 accuracy.
// Memory-bound (~0.66 GB total traffic -> ~28us at 23.3 TB/s); W stays L2-resident.

typedef __attribute__((ext_vector_type(16))) __bf16 v16bf;
typedef __attribute__((ext_vector_type(8)))  float v8f;

#define FEATN 256
#define LN_EPS 1e-5f

// split f32 into bf16 hi + bf16 lo(residual); x - hi is exact in f32.
__device__ __forceinline__ void split_bf16(float x, unsigned short& hi, unsigned short& lo) {
    unsigned u = __float_as_uint(x);
    unsigned short h = (unsigned short)(u >> 16);
    float fh = __uint_as_float(((unsigned)h) << 16);
    float r = x - fh;
    hi = h;
    lo = (unsigned short)(__float_as_uint(r) >> 16);
}

__device__ __forceinline__ v16bf make_frag(uint4 a, uint4 b) {
    union { uint4 u[2]; v16bf v; } c;
    c.u[0] = a; c.u[1] = b;
    return c.v;
}

// One-shot: split both 256x256 f32 weight matrices into bf16 hi/lo planes in scratch.
// ws layout (ushort): [0)o2n_hi [65536)o2n_lo [131072)n2o_hi [196608)n2o_lo
__global__ void convert_weights_kernel(const float* __restrict__ w_o2n,
                                       const float* __restrict__ w_n2o,
                                       unsigned short* __restrict__ ws) {
    int i = blockIdx.x * blockDim.x + threadIdx.x;
    if (i >= 2 * 65536) return;
    float x;
    unsigned short *ph, *pl;
    if (i < 65536) {
        x = w_o2n[i];
        ph = ws + i;          pl = ws + 65536 + i;
    } else {
        int j = i - 65536;
        x = w_n2o[j];
        ph = ws + 131072 + j; pl = ws + 196608 + j;
    }
    unsigned short h, l;
    split_bf16(x, h, l);
    *ph = h; *pl = l;
}

// MODE 0: down path. rows = b*16384 + n over 4*16384; A row = 0.5*(4 corner rows of H_orig)
// MODE 1: up path.   rows = b*65536 + o over 4*65536; A row = 0.5*H_down[b, parent(o)]
// Block = 128 threads (4 wave32) handles a 16-row x 256-col output tile.
template <int MODE>
__global__ __launch_bounds__(128)
void fused_gemm_ln_kernel(const float* __restrict__ Hsrc,
                          const unsigned short* __restrict__ Whi,
                          const unsigned short* __restrict__ Wlo,
                          const float* __restrict__ gamma,
                          const float* __restrict__ beta,
                          float* __restrict__ out) {
    __shared__ alignas(16) unsigned short sAhi[16][FEATN];
    __shared__ alignas(16) unsigned short sAlo[16][FEATN];
    __shared__ alignas(16) float          sOut[16][264];   // padded: kill bank conflicts

    const int  tid  = threadIdx.x;
    const long base = (long)blockIdx.x * 16;

    // ---- Phase 1: build A tile (16x256) as bf16 hi/lo in LDS, coalesced on features ----
    for (int r = 0; r < 16; ++r) {
        const long grow = base + r;
        if (MODE == 0) {
            const int b = (int)(grow >> 14);
            const int n = (int)(grow & 16383);
            const int i = n >> 7, j = n & 127;
            const long o00 = (long)b * 65536 + (long)i * 512 + (long)j * 2;
            const float* p0 = Hsrc + (o00      ) * 256;
            const float* p1 = Hsrc + (o00 + 1  ) * 256;
            const float* p2 = Hsrc + (o00 + 256) * 256;
            const float* p3 = Hsrc + (o00 + 257) * 256;
            #pragma unroll
            for (int s = 0; s < 2; ++s) {
                const int f = tid + s * 128;
                const float a = 0.5f * (p0[f] + p1[f] + p2[f] + p3[f]);
                unsigned short h, l;
                split_bf16(a, h, l);
                sAhi[r][f] = h; sAlo[r][f] = l;
            }
        } else {
            const int b  = (int)(grow >> 16);
            const int o  = (int)(grow & 65535);
            const int oi = o >> 8, oj = o & 255;
            const long srow = (long)b * 16384 + (long)(oi >> 1) * 128 + (oj >> 1);
            const float* p = Hsrc + srow * 256;
            #pragma unroll
            for (int s = 0; s < 2; ++s) {
                const int f = tid + s * 128;
                const float a = 0.5f * p[f];
                unsigned short h, l;
                split_bf16(a, h, l);
                sAhi[r][f] = h; sAlo[r][f] = l;
            }
        }
    }
    __syncthreads();

    // ---- Phase 2: WMMA. wave w owns N-tiles w*4..w*4+3; K = 256 in 8 steps of 32 ----
    const int wave  = tid >> 5;
    const int lane  = tid & 31;
    const int lrow  = lane & 15;            // A row (M) / W row (N) within tile
    const int khalf = (lane >> 4) * 8;      // lanes 16-31 take the upper K half

    v8f acc[4];
    #pragma unroll
    for (int t = 0; t < 4; ++t) acc[t] = v8f{};

    #pragma unroll
    for (int k0 = 0; k0 < FEATN; k0 += 32) {
        const int c0 = k0 + khalf;          // K {0..7} / {8..15}
        const int c1 = k0 + 16 + khalf;     // K {16..23} / {24..31}
        const v16bf ah = make_frag(*(const uint4*)&sAhi[lrow][c0],
                                   *(const uint4*)&sAhi[lrow][c1]);
        const v16bf al = make_frag(*(const uint4*)&sAlo[lrow][c0],
                                   *(const uint4*)&sAlo[lrow][c1]);
        #pragma unroll
        for (int t = 0; t < 4; ++t) {
            const int wrow = (wave * 4 + t) * 16 + lrow;       // B col n = W row n
            const unsigned short* bh = Whi + (long)wrow * FEATN;
            const unsigned short* bl = Wlo + (long)wrow * FEATN;
            const v16bf fbh = make_frag(*(const uint4*)(bh + c0), *(const uint4*)(bh + c1));
            const v16bf fbl = make_frag(*(const uint4*)(bl + c0), *(const uint4*)(bl + c1));
            // bf16x3: Ah*Bh + Ah*Bl + Al*Bh, f32 accumulate
            acc[t] = __builtin_amdgcn_wmma_f32_16x16x32_bf16(false, ah, false, fbh,
                                                             (short)0, acc[t], false, false);
            acc[t] = __builtin_amdgcn_wmma_f32_16x16x32_bf16(false, ah, false, fbl,
                                                             (short)0, acc[t], false, false);
            acc[t] = __builtin_amdgcn_wmma_f32_16x16x32_bf16(false, al, false, fbh,
                                                             (short)0, acc[t], false, false);
        }
    }

    // D layout: VGPR v holds D[v + 8*(lane>=16)][lane%16] within the 16x16 tile
    const int mhi = (lane >> 4) * 8;
    #pragma unroll
    for (int t = 0; t < 4; ++t) {
        const int col = (wave * 4 + t) * 16 + lrow;
        #pragma unroll
        for (int v = 0; v < 8; ++v)
            sOut[v + mhi][col] = acc[t][v];
    }
    __syncthreads();

    // ---- Phase 3: per-row LayerNorm + ReLU, one wave per row, coalesced f32 stores ----
    #pragma unroll
    for (int q = 0; q < 4; ++q) {
        const int r = wave + q * 4;
        float x[8], s = 0.f, ss = 0.f;
        #pragma unroll
        for (int i = 0; i < 8; ++i) {
            const float v = sOut[r][lane * 8 + i];
            x[i] = v; s += v; ss += v * v;
        }
        #pragma unroll
        for (int off = 16; off > 0; off >>= 1) {
            s  += __shfl_xor(s,  off, 32);
            ss += __shfl_xor(ss, off, 32);
        }
        const float mean = s * (1.0f / 256.0f);
        const float var  = ss * (1.0f / 256.0f) - mean * mean;   // population var, as jnp.var
        const float rstd = rsqrtf(var + LN_EPS);
        float* po = out + (base + r) * 256 + lane * 8;
        #pragma unroll
        for (int i = 0; i < 8; ++i) {
            const int f = lane * 8 + i;
            const float y = (x[i] - mean) * rstd * gamma[f] + beta[f];
            po[i] = fmaxf(y, 0.0f);
        }
    }
}

extern "C" void kernel_launch(void* const* d_in, const int* in_sizes, int n_in,
                              void* d_out, int out_size, void* d_ws, size_t ws_size,
                              hipStream_t stream) {
    const float* H_orig = (const float*)d_in[0];
    const float* H_down = (const float*)d_in[1];
    const float* W_o2n  = (const float*)d_in[2];
    const float* W_n2o  = (const float*)d_in[3];
    const float* g_o2n  = (const float*)d_in[4];
    const float* b_o2n  = (const float*)d_in[5];
    const float* g_n2o  = (const float*)d_in[6];
    const float* b_n2o  = (const float*)d_in[7];

    float* out = (float*)d_out;
    unsigned short* ws = (unsigned short*)d_ws;   // 512 KB used

    // Split weights to bf16 hi/lo once (L2-resident afterwards).
    convert_weights_kernel<<<512, 256, 0, stream>>>(W_o2n, W_n2o, ws);

    // Output order: H_orig_updated (4*65536*256 floats) then H_new_updated.
    // Up path: 262144 rows -> 16384 tiles.
    fused_gemm_ln_kernel<1><<<16384, 128, 0, stream>>>(
        H_down, ws + 131072, ws + 196608, g_n2o, b_n2o, out);
    // Down path: 65536 rows -> 4096 tiles.
    fused_gemm_ln_kernel<0><<<4096, 128, 0, stream>>>(
        H_orig, ws, ws + 65536, g_o2n, b_o2n, out + (long)4 * 65536 * 256);
}
